// ScaledDotProduct_77068893160143
// MI455X (gfx1250) — matrix operand order repaired and checked by reading, not compiled, not run on hardware
//
#include <hip/hip_runtime.h>

// ---------------------------------------------------------------------------
// Flash attention, B=2,H=8,S=4096,D=64 fp32 in/out, scale 1/sqrt(512).
// Round 3: transposed formulation  S^T = K Q^T,  O^T = V^T P^T.
// WMMA C-layout puts lane = query, so softmax max/sum are per-lane register
// reductions + ONE shfl_xor(16) each (no butterfly). f16 WMMA 16x16x32,
// TK=64 tiles, async-DMA LDS staging (global_load_async_to_lds_b128),
// K/V pre-converted to f16 (V pre-transposed) by prep kernels.
// ---------------------------------------------------------------------------

typedef __attribute__((ext_vector_type(16))) _Float16 v16h;
typedef __attribute__((ext_vector_type(8)))  _Float16 v8h;
typedef __attribute__((ext_vector_type(4)))  _Float16 v4h;
typedef __attribute__((ext_vector_type(8)))  float    v8f;

#define B_  2
#define H_  8
#define S_  4096
#define D_  64
#define QB  128            // query rows per block (8 waves x 16 rows)
#define TK  64             // keys per tile
#define NT  (S_ / TK)      // 64 key tiles
#define NW  (QB / 16)
#define NTHREADS (NW * 32)

#define KSTR 72            // f16 row stride (144B: 16B aligned, banks 36n%64 distinct)
#define VSTR 72
#define PSTR 72

#define WMMA_F16(A, B, C) \
  __builtin_amdgcn_wmma_f32_16x16x32_f16(false, (A), false, (B), (short)0, (C), false, false)

// B-operand / contiguous-16 loader: element e <-> K = base + hi*16 + e
__device__ __forceinline__ v16h ldv16(const _Float16* p) {
  const v8h* q = (const v8h*)p;
  v8h a = q[0], b = q[1];
  v16h r;
  #pragma unroll
  for (int i = 0; i < 8; ++i) { r[i] = a[i]; r[i + 8] = b[i]; }
  return r;
}

// A-operand loader from an LDS row: element e -> K=koff+hi*16+e (contiguous)
__device__ __forceinline__ v16h ldA(const _Float16* row, int koff, int hi) {
  const v8h c0 = *(const v8h*)(row + koff + hi * 16);
  const v8h c1 = *(const v8h*)(row + koff + hi * 16 + 8);
  v16h r;
  #pragma unroll
  for (int i = 0; i < 8; ++i) { r[i] = c0[i]; r[i + 8] = c1[i]; }
  return r;
}

// ---------------- prep kernel 1: K fp32 -> f16 (same layout) ----------------
__global__ __launch_bounds__(256)
void cvt_k_f16(const float* __restrict__ K, _Float16* __restrict__ Kh) {
  const size_t i = ((size_t)blockIdx.x * 256 + threadIdx.x) * 4;
  const float4 v = *(const float4*)(K + i);
  v4h h;
  h[0] = (_Float16)v.x; h[1] = (_Float16)v.y;
  h[2] = (_Float16)v.z; h[3] = (_Float16)v.w;
  *(v4h*)(Kh + i) = h;
}

// ------- prep kernel 2: V [B,S,H,D] fp32 -> Vt f16 [B*H][D][S] (transposed) -------
__global__ __launch_bounds__(256)
void tr_v_f16(const float* __restrict__ V, _Float16* __restrict__ Vt) {
  __shared__ _Float16 tile[D_][KSTR];
  const int bh = blockIdx.y;
  const int b0 = bh / H_, hh = bh % H_;
  const int s0 = blockIdx.x * 64;
  const int tid = threadIdx.x;
  #pragma unroll
  for (int i = 0; i < 4; ++i) {                  // 64x64 f32 = 1024 float4
    const int f  = tid + i * 256;
    const int sl = f >> 4;
    const int c4 = f & 15;
    const float4 vv = *(const float4*)(V + ((size_t)(b0 * S_ + s0 + sl) * H_ + hh) * D_ + c4 * 4);
    tile[c4 * 4 + 0][sl] = (_Float16)vv.x;
    tile[c4 * 4 + 1][sl] = (_Float16)vv.y;
    tile[c4 * 4 + 2][sl] = (_Float16)vv.z;
    tile[c4 * 4 + 3][sl] = (_Float16)vv.w;
  }
  __syncthreads();
  #pragma unroll
  for (int i = 0; i < 2; ++i) {                  // 64 d-rows x 8 chunks of 8 f16
    const int f  = tid + i * 256;
    const int d  = f >> 3;
    const int c8 = f & 7;
    const v8h h = *(const v8h*)&tile[d][c8 * 8];
    *(v8h*)(Vt + ((size_t)bh * D_ + d) * S_ + s0 + c8 * 8) = h;
  }
}

// ------------------------------ main kernel ------------------------------
__global__ __launch_bounds__(NTHREADS)
void fattn_f16wmma(const float* __restrict__ Q, const _Float16* __restrict__ Kh,
                   const _Float16* __restrict__ Vth, float* __restrict__ O) {
  __shared__ _Float16 kb[2][TK][KSTR];   // K tile   f16 [key][d]
  __shared__ _Float16 vt[2][D_][VSTR];   // V^T tile f16 [d][key]
  __shared__ _Float16 pb[NW][16][PSTR];  // per-wave P scratch [query][key]

  const int tid  = threadIdx.x;
  const int w    = tid >> 5;
  const int lane = tid & 31;
  const int hi   = lane >> 4;
  const int nn   = lane & 15;

  const int bh = blockIdx.y;
  const int q0 = blockIdx.x * QB;

  const float*    Qg  = Q   + (size_t)bh * S_ * D_;
  const _Float16* Kg  = Kh  + (size_t)bh * S_ * D_;
  const _Float16* Vtg = Vth + (size_t)bh * D_ * S_;
  float*          Og  = O   + (size_t)bh * S_ * D_;

  // fold 1/sqrt(512) and log2(e) into Q so softmax uses exp2
  const float qs = 0.04419417382415922f * 1.4426950408889634f;

  // Q^T as B-operands: lane = query column nn; K(d) = hi*16+e (contiguous)
  const float* qp = Qg + (size_t)(q0 + w * 16 + nn) * D_;
  v16h qb_lo, qb_hi;
  #pragma unroll
  for (int e = 0; e < 16; ++e) {
    qb_lo[e] = (_Float16)(qp[     hi * 16 + e] * qs);
    qb_hi[e] = (_Float16)(qp[32 + hi * 16 + e] * qs);
  }

  v8f acc0 = {}, acc1 = {}, acc2 = {}, acc3 = {};  // O^T accumulators (C layout)
  float mrun = -__builtin_inff();                  // per-lane (= per-query) stats
  float lrun = 0.0f;

  // async DMA staging: 512 x 16B chunks each for K tile and V^T tile
  auto stage = [&](int tt, int buf) {
    #pragma unroll
    for (int i = 0; i < 2; ++i) {
      const int f  = tid + i * NTHREADS;   // 0..511
      const int rr = f >> 3;               // row (key for K, d for V^T)
      const int c  = f & 7;                // 16B chunk in 128B row
      {
        unsigned lds = (unsigned)(uintptr_t)&kb[buf][rr][c * 8];
        unsigned long long ga =
            (unsigned long long)(uintptr_t)(Kg + (size_t)(tt * TK + rr) * D_ + c * 8);
        asm volatile("global_load_async_to_lds_b128 %0, %1, off"
                     :: "v"(lds), "v"(ga) : "memory");
      }
      {
        unsigned lds = (unsigned)(uintptr_t)&vt[buf][rr][c * 8];
        unsigned long long ga =
            (unsigned long long)(uintptr_t)(Vtg + (size_t)rr * S_ + tt * TK + c * 8);
        asm volatile("global_load_async_to_lds_b128 %0, %1, off"
                     :: "v"(lds), "v"(ga) : "memory");
      }
    }
  };

  stage(0, 0);
  asm volatile("s_wait_asynccnt 0x0" ::: "memory");
  __syncthreads();

  for (int t = 0; t < NT; ++t) {
    const int cur = t & 1;
    if (t + 1 < NT) stage(t + 1, cur ^ 1);   // DMA next tile during compute

    // ---- S^T = K Q^T : four 16-key tiles (M=key, N=query), K-depth = d = 64 ----
    const _Float16 (*kbc)[KSTR] = kb[cur];
    v8f st0 = {}, st1 = {}, st2 = {}, st3 = {};
    {
      v16h a0 = ldA(kbc[ 0 + nn], 0, hi), a1 = ldA(kbc[ 0 + nn], 32, hi);
      st0 = WMMA_F16(a0, qb_lo, st0); st0 = WMMA_F16(a1, qb_hi, st0);
      v16h a2 = ldA(kbc[16 + nn], 0, hi), a3 = ldA(kbc[16 + nn], 32, hi);
      st1 = WMMA_F16(a2, qb_lo, st1); st1 = WMMA_F16(a3, qb_hi, st1);
      v16h a4 = ldA(kbc[32 + nn], 0, hi), a5 = ldA(kbc[32 + nn], 32, hi);
      st2 = WMMA_F16(a4, qb_lo, st2); st2 = WMMA_F16(a5, qb_hi, st2);
      v16h a6 = ldA(kbc[48 + nn], 0, hi), a7 = ldA(kbc[48 + nn], 32, hi);
      st3 = WMMA_F16(a6, qb_lo, st3); st3 = WMMA_F16(a7, qb_hi, st3);
    }

    // ---- per-lane softmax: lane owns query nn; keys r+8*hi+16*mt ----
    float vm = st0[0];
    #pragma unroll
    for (int r = 0; r < 8; ++r) {
      vm = fmaxf(vm, st0[r]); vm = fmaxf(vm, st1[r]);
      vm = fmaxf(vm, st2[r]); vm = fmaxf(vm, st3[r]);
    }
    vm = fmaxf(vm, __shfl_xor(vm, 16, 32));        // merge the two key-halves
    const float mn = fmaxf(mrun, vm);
    const float al = exp2f(mrun - mn);
    mrun = mn;

    v8h p0, p1, p2, p3;
    float lsum = 0.0f;
    #pragma unroll
    for (int r = 0; r < 8; ++r) {
      const float x0 = exp2f(st0[r] - mn); p0[r] = (_Float16)x0;
      const float x1 = exp2f(st1[r] - mn); p1[r] = (_Float16)x1;
      const float x2 = exp2f(st2[r] - mn); p2[r] = (_Float16)x2;
      const float x3 = exp2f(st3[r] - mn); p3[r] = (_Float16)x3;
      lsum += (x0 + x1) + (x2 + x3);
    }
    lsum += __shfl_xor(lsum, 16, 32);
    lrun = lrun * al + lsum;
    #pragma unroll
    for (int r = 0; r < 8; ++r) {
      acc0[r] *= al; acc1[r] *= al; acc2[r] *= al; acc3[r] *= al;
    }

    // P^T (C layout) -> pb[query][key]: contiguous b128 stores per key-tile
    *(v8h*)&pb[w][nn][ 0 + 8 * hi] = p0;
    *(v8h*)&pb[w][nn][16 + 8 * hi] = p1;
    *(v8h*)&pb[w][nn][32 + 8 * hi] = p2;
    *(v8h*)&pb[w][nn][48 + 8 * hi] = p3;

    // ---- O^T += V^T P^T : A = V^T d-tiles (from vt), B = P^T (from pb) ----
    {
      v16h pblo = ldv16(&pb[w][nn][     hi * 16]);   // keys  0..31
      v16h pbhi = ldv16(&pb[w][nn][32 + hi * 16]);   // keys 32..63
      const _Float16 (*vtc)[VSTR] = vt[cur];
      v16h v0 = ldA(vtc[ 0 + nn], 0, hi), v1 = ldA(vtc[ 0 + nn], 32, hi);
      acc0 = WMMA_F16(v0, pblo, acc0); acc0 = WMMA_F16(v1, pbhi, acc0);
      v16h v2 = ldA(vtc[16 + nn], 0, hi), v3 = ldA(vtc[16 + nn], 32, hi);
      acc1 = WMMA_F16(v2, pblo, acc1); acc1 = WMMA_F16(v3, pbhi, acc1);
      v16h v4 = ldA(vtc[32 + nn], 0, hi), v5 = ldA(vtc[32 + nn], 32, hi);
      acc2 = WMMA_F16(v4, pblo, acc2); acc2 = WMMA_F16(v5, pbhi, acc2);
      v16h v6 = ldA(vtc[48 + nn], 0, hi), v7 = ldA(vtc[48 + nn], 32, hi);
      acc3 = WMMA_F16(v6, pblo, acc3); acc3 = WMMA_F16(v7, pbhi, acc3);
    }

    asm volatile("s_wait_asynccnt 0x0" ::: "memory");  // own DMA for t+1 landed
    __syncthreads();                                   // everyone's DMA landed
  }

  // ---- epilogue: lane owns output row (q0+w*16+nn); d = dt*16 + 8*hi + r ----
  const float inv = 1.0f / lrun;
  float* op = Og + (size_t)(q0 + w * 16 + nn) * D_ + 8 * hi;
  {
    float4 lo, hii;
    lo.x = acc0[0] * inv; lo.y = acc0[1] * inv; lo.z = acc0[2] * inv; lo.w = acc0[3] * inv;
    hii.x = acc0[4] * inv; hii.y = acc0[5] * inv; hii.z = acc0[6] * inv; hii.w = acc0[7] * inv;
    *(float4*)(op +  0) = lo; *(float4*)(op +  4) = hii;
    lo.x = acc1[0] * inv; lo.y = acc1[1] * inv; lo.z = acc1[2] * inv; lo.w = acc1[3] * inv;
    hii.x = acc1[4] * inv; hii.y = acc1[5] * inv; hii.z = acc1[6] * inv; hii.w = acc1[7] * inv;
    *(float4*)(op + 16) = lo; *(float4*)(op + 20) = hii;
    lo.x = acc2[0] * inv; lo.y = acc2[1] * inv; lo.z = acc2[2] * inv; lo.w = acc2[3] * inv;
    hii.x = acc2[4] * inv; hii.y = acc2[5] * inv; hii.z = acc2[6] * inv; hii.w = acc2[7] * inv;
    *(float4*)(op + 32) = lo; *(float4*)(op + 36) = hii;
    lo.x = acc3[0] * inv; lo.y = acc3[1] * inv; lo.z = acc3[2] * inv; lo.w = acc3[3] * inv;
    hii.x = acc3[4] * inv; hii.y = acc3[5] * inv; hii.z = acc3[6] * inv; hii.w = acc3[7] * inv;
    *(float4*)(op + 48) = lo; *(float4*)(op + 52) = hii;
  }
}

extern "C" void kernel_launch(void* const* d_in, const int* in_sizes, int n_in,
                              void* d_out, int out_size, void* d_ws, size_t ws_size,
                              hipStream_t stream) {
  (void)in_sizes; (void)n_in; (void)out_size; (void)ws_size;
  const float* q = (const float*)d_in[0];
  const float* k = (const float*)d_in[1];
  const float* v = (const float*)d_in[2];
  float*       o = (float*)d_out;

  const size_t elems = (size_t)B_ * H_ * S_ * D_;       // 4.19M
  _Float16* Kh  = (_Float16*)d_ws;                       // 8 MB
  _Float16* Vth = (_Float16*)((char*)d_ws + elems * 2);  // 8 MB

  cvt_k_f16<<<dim3((unsigned)(elems / 1024)), 256, 0, stream>>>(k, Kh);
  tr_v_f16 <<<dim3(S_ / 64, B_ * H_), 256, 0, stream>>>(v, Vth);
  fattn_f16wmma<<<dim3(S_ / QB, B_ * H_), NTHREADS, 0, stream>>>(q, Kh, Vth, o);
}
